// iKT3_82738249990697
// MI455X (gfx1250) — matrix operand order repaired and checked by reading, not compiled, not run on hardware
//
#include <hip/hip_runtime.h>

// ---------------------------------------------------------------------------
// Problem constants (match reference)
// ---------------------------------------------------------------------------
#define BDIM   8
#define LSEQ   1024
#define DMODEL 512
#define NHEAD  8
#define DKH    64
#define DFFN   2048
#define MROWS  (BDIM * LSEQ)   // 8192 token rows
#define NUMC   1000

// ---------------------------------------------------------------------------
// WMMA types / helpers (gfx1250, wave32)
// ---------------------------------------------------------------------------
typedef __attribute__((ext_vector_type(16))) __bf16 v16bf;
typedef __attribute__((ext_vector_type(8)))  float  v8f;

union Frag {
    v16bf          v;
    unsigned int   w[8];
    unsigned short u[16];
};

static __device__ __forceinline__ unsigned short f32_to_bf16(float f) {
    unsigned int x = __float_as_uint(f);
    x += 0x7fffu + ((x >> 16) & 1u);          // round-to-nearest-even
    return (unsigned short)(x >> 16);
}

static __device__ __forceinline__ v8f wmma_bf16(v16bf a, v16bf b, v8f c) {
    // D = A(16x32 bf16) * B(32x16 bf16) + C(16x16 f32)
    return __builtin_amdgcn_wmma_f32_16x16x32_bf16(false, a, false, b,
                                                   (short)0, c, false, false);
}

// low 32 bits of a generic pointer to LDS == LDS byte offset (aperture layout)
static __device__ __forceinline__ unsigned int lds_addr(const void* p) {
    return (unsigned int)(size_t)p;
}

// ---------------------------------------------------------------------------
// f32 [batch][K][N]  ->  bf16 [batch][N][K]  (transposed weight conversion)
// ---------------------------------------------------------------------------
__global__ void cvt_t_kernel(const float* __restrict__ src,
                             unsigned short* __restrict__ dst, int K, int N) {
    const size_t base = (size_t)blockIdx.y * K * N;
    const int i = blockIdx.x * 256 + threadIdx.x;
    if (i < K * N) {
        const int k = i / N, n = i - k * N;
        dst[base + (size_t)n * K + k] = f32_to_bf16(src[base + i]);
    }
}

// ---------------------------------------------------------------------------
// Embedding gather: ctx/val = emb[q + NUM_C*r] + pos  (f32 master + bf16 copy)
// ---------------------------------------------------------------------------
__global__ __launch_bounds__(256)
void embed_kernel(const int* __restrict__ q, const int* __restrict__ r,
                  const float* __restrict__ ctx_emb, const float* __restrict__ val_emb,
                  const float* __restrict__ pos_emb,
                  float* __restrict__ ctxf, unsigned short* __restrict__ ctxb,
                  float* __restrict__ valf, unsigned short* __restrict__ valb) {
    const int row = blockIdx.x;            // b*L + l
    const int l   = row & (LSEQ - 1);
    const int it  = q[row] + NUMC * r[row];
    for (int d = threadIdx.x; d < DMODEL; d += 256) {
        float p = pos_emb[(size_t)l * DMODEL + d];
        float c = ctx_emb[(size_t)it * DMODEL + d] + p;
        float v = val_emb[(size_t)it * DMODEL + d] + p;
        size_t o = (size_t)row * DMODEL + d;
        ctxf[o] = c; ctxb[o] = f32_to_bf16(c);
        valf[o] = v; valb[o] = f32_to_bf16(v);
    }
}

// ---------------------------------------------------------------------------
// WMMA GEMM:  C[M,N] = act(A_bf16[M,K] @ BT_bf16[N,K]^T + bias[N])
// 128x128 tile / 256-thread WG. 8 waves, wave w owns row strip w (16 rows) and
// all 8 col strips -> 8 accumulators, 8 WMMAs per 32-K step.
// Double-buffered LDS filled by GLOBAL_LOAD_ASYNC_TO_LDS_B128: the k+32 tile's
// DMA overlaps the current tile's WMMAs; one barrier per K-step.
//   RAW: each wave does s_wait_asynccnt 0 BEFORE the barrier, so passing the
//        barrier proves every wave's current-tile DMA has landed.
//   WAR: next-tile DMA is issued only after the barrier that proves all waves
//        finished reading that buffer in the previous round.
// All 8 B fragments are materialized before the WMMA chain so the scheduler
// can stage s_wait_dscnt instead of a full wait per WMMA.
// M % 128 == 0, N % 128 == 0, K % 32 == 0 at every call site.
// ---------------------------------------------------------------------------
__global__ __launch_bounds__(256)
void wmma_gemm_kernel(const unsigned short* __restrict__ A,   // [M][K]
                      const unsigned short* __restrict__ BT,  // [N][K] (pre-transposed)
                      const float* __restrict__ bias,
                      float* __restrict__ Cf,
                      unsigned short* __restrict__ Cb,
                      int M, int N, int K, int act) {
    __shared__ unsigned short As[2][128 * 40];   // 128 rows x 32 K, stride 40
    __shared__ unsigned short Bs[2][128 * 40];   // 128 cols x 32 K

    const int tid  = threadIdx.x;
    const int lane = tid & 31;
    const int wave = tid >> 5;               // row strip 0..7
    const int ln   = lane & 15;
    const int kh   = lane >> 4;
    const int m0   = blockIdx.y * 128;
    const int n0   = blockIdx.x * 128;

    const int lrow = tid >> 1;               // 0..127
    const int lcg  = (tid & 1) * 16;         // ushort col group 0 / 16

    const unsigned short* gA = A  + (size_t)(m0 + lrow) * K + lcg;
    const unsigned short* gB = BT + (size_t)(n0 + lrow) * K + lcg;
    const unsigned int lA0 = lds_addr(As[0] + lrow * 40 + lcg);
    const unsigned int lA1 = lds_addr(As[1] + lrow * 40 + lcg);
    const unsigned int lB0 = lds_addr(Bs[0] + lrow * 40 + lcg);
    const unsigned int lB1 = lds_addr(Bs[1] + lrow * 40 + lcg);

    auto issue = [&](int buf, int k0) {
        const unsigned int la = buf ? lA1 : lA0;
        const unsigned int lb = buf ? lB1 : lB0;
        asm volatile("global_load_async_to_lds_b128 %0, %1, off"
                     :: "v"(la), "v"(gA + k0) : "memory");
        asm volatile("global_load_async_to_lds_b128 %0, %1, off"
                     :: "v"(la + 16u), "v"(gA + k0 + 8) : "memory");
        asm volatile("global_load_async_to_lds_b128 %0, %1, off"
                     :: "v"(lb), "v"(gB + k0) : "memory");
        asm volatile("global_load_async_to_lds_b128 %0, %1, off"
                     :: "v"(lb + 16u), "v"(gB + k0 + 8) : "memory");
    };

    v8f acc[8] = {};
    issue(0, 0);                              // prologue DMA for first tile

    for (int k0 = 0; k0 < K; k0 += 32) {
        const int cur = (k0 >> 5) & 1;
        asm volatile("s_wait_asynccnt 0x0" ::: "memory");   // own DMA landed
        __syncthreads();                                    // everyone's DMA landed
        if (k0 + 32 < K) {
            issue(cur ^ 1, k0 + 32);          // next tile DMA overlaps compute
            if (k0 + 64 < K) {
                __builtin_prefetch(gA + k0 + 64, 0, 1);
                __builtin_prefetch(gB + k0 + 64, 0, 1);
            }
        }
        const unsigned short* as = As[cur];
        const unsigned short* bs = Bs[cur];

        Frag fa;   // A frag: K = kh*16 + i
        {
            const unsigned int* p  = (const unsigned int*)(as + (wave * 16 + ln) * 40 + kh * 16);
            const unsigned int* p2 = (const unsigned int*)(as + (wave * 16 + ln) * 40 + kh * 16 + 8);
            fa.w[0] = p[0];  fa.w[1] = p[1];  fa.w[2] = p[2];  fa.w[3] = p[3];
            fa.w[4] = p2[0]; fa.w[5] = p2[1]; fa.w[6] = p2[2]; fa.w[7] = p2[3];
        }
        Frag fb[8];   // all B frags live -> staged DS waits, pipelined WMMAs
        #pragma unroll
        for (int c = 0; c < 8; ++c) {
            const unsigned int* p = (const unsigned int*)(bs + (c * 16 + ln) * 40 + kh * 16);
            #pragma unroll
            for (int j = 0; j < 8; ++j) fb[c].w[j] = p[j];
        }
        #pragma unroll
        for (int c = 0; c < 8; ++c) acc[c] = wmma_bf16(fa.v, fb[c].v, acc[c]);
        // no trailing barrier: next iteration's barrier orders reuse of 'cur'
    }

    // Epilogue: C layout -> row = wave*16 + kh*8 + r, col = c*16 + ln
    const int mb = m0 + wave * 16 + kh * 8;
    #pragma unroll
    for (int c = 0; c < 8; ++c) {
        const int n = n0 + c * 16 + ln;
        const float bn = bias ? bias[n] : 0.0f;
        #pragma unroll
        for (int r = 0; r < 8; ++r) {
            float v = acc[c][r] + bn;
            if (act == 1) v = fmaxf(v, 0.0f);
            size_t o = (size_t)(mb + r) * N + n;
            if (Cf) Cf[o] = v;
            if (Cb) Cb[o] = f32_to_bf16(v);
        }
    }
}

// ---------------------------------------------------------------------------
// V transpose: [b*L, D] bf16 -> [b, h, dk, L] bf16 so P.V fragments are b128
// ---------------------------------------------------------------------------
__global__ __launch_bounds__(256)
void vtrans_kernel(const unsigned short* __restrict__ v, unsigned short* __restrict__ vt) {
    const int row = blockIdx.x;              // b*L + l
    const int b = row >> 10, l = row & (LSEQ - 1);
    for (int d = threadIdx.x; d < DMODEL; d += 256)
        vt[((size_t)(b * NHEAD + (d >> 6)) * DKH + (d & 63)) * LSEQ + l] =
            v[(size_t)row * DMODEL + d];
}

// ---------------------------------------------------------------------------
// Flash attention (causal), per (b, h, 64-row q block), 4 waves x 16 q rows.
// Q.K^T and P.V via bf16 WMMA; online softmax; P relaid via per-wave LDS tile.
// Q/K: [B*L, D] bf16 (head h = cols h*64..);  V: transposed [b,h,dk,L] bf16.
// ---------------------------------------------------------------------------
__global__ __launch_bounds__(128)
void attn_kernel(const unsigned short* __restrict__ Qb,
                 const unsigned short* __restrict__ Kb,
                 const unsigned short* __restrict__ VT,
                 unsigned short* __restrict__ Ob) {
    __shared__ unsigned short Pls[4 * 16 * 40];   // per-wave 16x32 prob tile (stride 40)

    const int tid  = threadIdx.x;
    const int lane = tid & 31;
    const int wave = tid >> 5;
    const int ln   = lane & 15;
    const int kh   = lane >> 4;
    const int bq   = blockIdx.x;
    const int h    = blockIdx.y;
    const int b    = blockIdx.z;
    const int qr0  = bq * 64 + wave * 16;
    const size_t base  = (size_t)b * LSEQ * DMODEL;
    const size_t vbase = (size_t)(b * NHEAD + h) * DKH * LSEQ;
    const int hoff = h * DKH;

    // Q fragments (constant over kv loop): A layout, dk steps {0,32}
    Frag fq[2];
    {
        const unsigned short* qrow = Qb + base + (size_t)(qr0 + ln) * DMODEL + hoff;
        #pragma unroll
        for (int s = 0; s < 2; ++s) {
            const unsigned int* p  = (const unsigned int*)(qrow + s * 32 + kh * 16);
            const unsigned int* p2 = (const unsigned int*)(qrow + s * 32 + kh * 16 + 8);
            fq[s].w[0] = p[0];  fq[s].w[1] = p[1];  fq[s].w[2] = p[2];  fq[s].w[3] = p[3];
            fq[s].w[4] = p2[0]; fq[s].w[5] = p2[1]; fq[s].w[6] = p2[2]; fq[s].w[7] = p2[3];
        }
    }

    v8f o[4] = {};                 // output accum, 4 dk column groups of 16
    float mr[8], lr[8];            // per-row running max / sum (replicated in half)
    #pragma unroll
    for (int r = 0; r < 8; ++r) { mr[r] = -1e30f; lr[r] = 0.0f; }

    for (int kt = 0; kt < qr0 + 16; kt += 32) {
        #pragma unroll
        for (int st = 0; st < 2; ++st) {
            const int kc = kt + st * 16;
            if (kc >= qr0 + 16) {          // fully-masked subtile: zero probs
                #pragma unroll
                for (int r = 0; r < 8; ++r)
                    Pls[(wave * 16 + kh * 8 + r) * 40 + st * 16 + ln] = 0;
                continue;
            }
            // S = Q . K^T  (accumulate over dk in two 32-steps)
            v8f s_frag = {};
            #pragma unroll
            for (int d = 0; d < 2; ++d) {
                Frag fk;   // B layout: K = kh*16 + i (contiguous dk in memory)
                const unsigned int* p = (const unsigned int*)
                    (Kb + base + (size_t)(kc + ln) * DMODEL + hoff + d * 32 + kh * 16);
                #pragma unroll
                for (int j = 0; j < 8; ++j) fk.w[j] = p[j];
                s_frag = wmma_bf16(fq[d].v, fk.v, s_frag);
            }
            // scale, causal mask, online softmax update
            const int kpos = kc + ln;
            #pragma unroll
            for (int r = 0; r < 8; ++r) {
                const int qpos = qr0 + kh * 8 + r;
                float sv = s_frag[r] * 0.125f;          // 1/sqrt(64)
                if (kpos > qpos) sv = -1e30f;
                float rm = sv;
                rm = fmaxf(rm, __shfl_xor(rm, 1));
                rm = fmaxf(rm, __shfl_xor(rm, 2));
                rm = fmaxf(rm, __shfl_xor(rm, 4));
                rm = fmaxf(rm, __shfl_xor(rm, 8));
                const float mnew  = fmaxf(mr[r], rm);
                const float alpha = __expf(mr[r] - mnew);
                const float pv    = __expf(sv - mnew);
                float rs = pv;
                rs += __shfl_xor(rs, 1);
                rs += __shfl_xor(rs, 2);
                rs += __shfl_xor(rs, 4);
                rs += __shfl_xor(rs, 8);
                lr[r] = lr[r] * alpha + rs;
                mr[r] = mnew;
                #pragma unroll
                for (int f = 0; f < 4; ++f) o[f][r] *= alpha;
                Pls[(wave * 16 + kh * 8 + r) * 40 + st * 16 + ln] = f32_to_bf16(pv);
            }
        }
        // same-wave LDS RAW: drain DS counter before re-reading P in A layout
        asm volatile("s_wait_dscnt 0x0" ::: "memory");

        Frag fp;   // P as A fragment (16 q rows x 32 keys)
        {
            const unsigned int* p  = (const unsigned int*)(Pls + (wave * 16 + ln) * 40 + kh * 16);
            const unsigned int* p2 = (const unsigned int*)(Pls + (wave * 16 + ln) * 40 + kh * 16 + 8);
            fp.w[0] = p[0];  fp.w[1] = p[1];  fp.w[2] = p[2];  fp.w[3] = p[3];
            fp.w[4] = p2[0]; fp.w[5] = p2[1]; fp.w[6] = p2[2]; fp.w[7] = p2[3];
        }
        #pragma unroll
        for (int g = 0; g < 4; ++g) {
            Frag fv;   // V as B fragment: K = key pos (contiguous in VT), N = dk col
            const unsigned int* p = (const unsigned int*)
                (VT + vbase + (size_t)(g * 16 + ln) * LSEQ + kt + kh * 16);
            #pragma unroll
            for (int j = 0; j < 8; ++j) fv.w[j] = p[j];
            o[g] = wmma_bf16(fp.v, fv.v, o[g]);
        }
    }

    // normalize and store (bf16, feeds the Wo GEMM)
    #pragma unroll
    for (int g = 0; g < 4; ++g) {
        #pragma unroll
        for (int r = 0; r < 8; ++r) {
            float v = o[g][r] / lr[r];
            Ob[base + (size_t)(qr0 + kh * 8 + r) * DMODEL + hoff + g * 16 + ln] = f32_to_bf16(v);
        }
    }
}

// ---------------------------------------------------------------------------
// y = LN(x + res) * s + b  -> f32 master + bf16 copy. One 256-thread WG / row.
// ---------------------------------------------------------------------------
__global__ __launch_bounds__(256)
void ln_residual_kernel(const float* __restrict__ x, const float* __restrict__ res,
                        const float* __restrict__ s, const float* __restrict__ bb,
                        float* __restrict__ outf, unsigned short* __restrict__ outb) {
    __shared__ float red[8];
    const int row  = blockIdx.x;
    const int tid  = threadIdx.x;
    const int lane = tid & 31;
    const int wv   = tid >> 5;
    const size_t o0 = (size_t)row * DMODEL + tid;
    const size_t o1 = o0 + 256;

    float v0 = x[o0] + res[o0];
    float v1 = x[o1] + res[o1];

    float ss = v0 + v1;
    ss += __shfl_xor(ss, 16); ss += __shfl_xor(ss, 8);
    ss += __shfl_xor(ss, 4);  ss += __shfl_xor(ss, 2); ss += __shfl_xor(ss, 1);
    if (lane == 0) red[wv] = ss;
    __syncthreads();
    float mu = 0.0f;
    #pragma unroll
    for (int i = 0; i < 8; ++i) mu += red[i];
    mu *= (1.0f / DMODEL);
    __syncthreads();

    const float d0 = v0 - mu, d1 = v1 - mu;
    float vs = d0 * d0 + d1 * d1;
    vs += __shfl_xor(vs, 16); vs += __shfl_xor(vs, 8);
    vs += __shfl_xor(vs, 4);  vs += __shfl_xor(vs, 2); vs += __shfl_xor(vs, 1);
    if (lane == 0) red[wv] = vs;
    __syncthreads();
    float var = 0.0f;
    #pragma unroll
    for (int i = 0; i < 8; ++i) var += red[i];
    var *= (1.0f / DMODEL);

    const float inv = rsqrtf(var + 1e-5f);
    const float y0 = d0 * inv * s[tid] + bb[tid];
    const float y1 = d1 * inv * s[tid + 256] + bb[tid + 256];
    outf[o0] = y0; outb[o0] = f32_to_bf16(y0);
    outf[o1] = y1; outb[o1] = f32_to_bf16(y1);
}

// ---------------------------------------------------------------------------
// feat = concat(ctx, val, skill_emb[q]) as bf16 [8192, 1536]
// ---------------------------------------------------------------------------
__global__ __launch_bounds__(256)
void feat_kernel(const float* __restrict__ ctxf, const float* __restrict__ valf,
                 const float* __restrict__ skill_emb, const int* __restrict__ q,
                 unsigned short* __restrict__ feat) {
    const int row = blockIdx.x;
    const int sk  = q[row];
    for (int d = threadIdx.x; d < DMODEL; d += 256) {
        feat[(size_t)row * 1536 + d]        = f32_to_bf16(ctxf[(size_t)row * DMODEL + d]);
        feat[(size_t)row * 1536 + 512 + d]  = f32_to_bf16(valf[(size_t)row * DMODEL + d]);
        feat[(size_t)row * 1536 + 1024 + d] = f32_to_bf16(skill_emb[(size_t)sk * DMODEL + d]);
    }
}

// ---------------------------------------------------------------------------
// out[row] = sigmoid(h2[row,:256] . hW3 + hb3)  — one wave per row
// ---------------------------------------------------------------------------
__global__ __launch_bounds__(256)
void head_final_kernel(const float* __restrict__ h2, const float* __restrict__ hW3,
                       const float* __restrict__ hb3, float* __restrict__ out) {
    const int row  = blockIdx.x * 8 + (threadIdx.x >> 5);
    const int lane = threadIdx.x & 31;
    float sum = 0.0f;
    #pragma unroll
    for (int j = lane; j < 256; j += 32) sum += h2[(size_t)row * 256 + j] * hW3[j];
    sum += __shfl_xor(sum, 16); sum += __shfl_xor(sum, 8);
    sum += __shfl_xor(sum, 4);  sum += __shfl_xor(sum, 2); sum += __shfl_xor(sum, 1);
    if (lane == 0) out[row] = 1.0f / (1.0f + __expf(-(sum + hb3[0])));
}

// ---------------------------------------------------------------------------
// Host orchestration
// ---------------------------------------------------------------------------
extern "C" void kernel_launch(void* const* d_in, const int* in_sizes, int n_in,
                              void* d_out, int out_size, void* d_ws, size_t ws_size,
                              hipStream_t stream) {
    const int*   q_ids   = (const int*)  d_in[0];
    const int*   r_ids   = (const int*)  d_in[1];
    const float* ctx_emb = (const float*)d_in[2];
    const float* val_emb = (const float*)d_in[3];
    const float* skill   = (const float*)d_in[4];
    const float* pos_emb = (const float*)d_in[5];
    const float* Wq = (const float*)d_in[6];   const float* bq = (const float*)d_in[7];
    const float* Wk = (const float*)d_in[8];   const float* bk = (const float*)d_in[9];
    const float* Wv = (const float*)d_in[10];  const float* bv = (const float*)d_in[11];
    const float* Wo = (const float*)d_in[12];  const float* bo = (const float*)d_in[13];
    const float* ln1c_s = (const float*)d_in[14]; const float* ln1c_b = (const float*)d_in[15];
    const float* ln1v_s = (const float*)d_in[16]; const float* ln1v_b = (const float*)d_in[17];
    const float* ln2c_s = (const float*)d_in[18]; const float* ln2c_b = (const float*)d_in[19];
    const float* ln2v_s = (const float*)d_in[20]; const float* ln2v_b = (const float*)d_in[21];
    const float* fc1cW = (const float*)d_in[22]; const float* fc1cb = (const float*)d_in[23];
    const float* fc2cW = (const float*)d_in[24]; const float* fc2cb = (const float*)d_in[25];
    const float* fc1vW = (const float*)d_in[26]; const float* fc1vb = (const float*)d_in[27];
    const float* fc2vW = (const float*)d_in[28]; const float* fc2vb = (const float*)d_in[29];
    const float* hW1 = (const float*)d_in[30]; const float* hb1 = (const float*)d_in[31];
    const float* hW2 = (const float*)d_in[32]; const float* hb2 = (const float*)d_in[33];
    const float* hW3 = (const float*)d_in[34]; const float* hb3 = (const float*)d_in[35];

    // ---- workspace carve-up (256B aligned) ----
    char* ws = (char*)d_ws;
    size_t off = 0;
    auto take = [&](size_t bytes) -> char* {
        char* p = ws + off;
        off += (bytes + 255) & ~(size_t)255;
        return p;
    };
    // bf16 transposed weight blob offsets (elements)
    const size_t oWq = 0, oWk = 1048576, oWv = 2097152, oWo = 3145728;
    const size_t oF1c = 4194304, oF2c = 8388608, oF1v = 12582912, oF2v = 16777216;
    const size_t oH1 = 20971520, oH2 = 24117248, wTot = 24641536;

    unsigned short* wbf  = (unsigned short*)take(wTot * 2);
    float*          ctxf = (float*)take((size_t)MROWS * DMODEL * 4);
    float*          valf = (float*)take((size_t)MROWS * DMODEL * 4);
    unsigned short* ctxb = (unsigned short*)take((size_t)MROWS * DMODEL * 2);
    unsigned short* valb = (unsigned short*)take((size_t)MROWS * DMODEL * 2);
    unsigned short* qb   = (unsigned short*)take((size_t)MROWS * DMODEL * 2);
    unsigned short* kb   = (unsigned short*)take((size_t)MROWS * DMODEL * 2);
    unsigned short* vb   = (unsigned short*)take((size_t)MROWS * DMODEL * 2);
    unsigned short* vt   = (unsigned short*)take((size_t)MROWS * DMODEL * 2);
    unsigned short* attb = (unsigned short*)take((size_t)MROWS * DMODEL * 2);
    float*          tmpf = (float*)take((size_t)MROWS * DMODEL * 4);
    unsigned short* bigb = (unsigned short*)take((size_t)MROWS * DFFN * 2);
    unsigned short* featb= (unsigned short*)take((size_t)MROWS * 1536 * 2);
    float*          h2f  = (float*)take((size_t)MROWS * 256 * 4);

    // convert + transpose weights: [batch][K][N] f32 -> [batch][N][K] bf16
    auto cvtT = [&](const float* src, unsigned short* dst, int K, int N, int batch) {
        cvt_t_kernel<<<dim3((unsigned)((K * N + 255) / 256), batch), 256, 0, stream>>>(src, dst, K, N);
    };
    cvtT(Wq, wbf + oWq, DMODEL, DMODEL, 4);
    cvtT(Wk, wbf + oWk, DMODEL, DMODEL, 4);
    cvtT(Wv, wbf + oWv, DMODEL, DMODEL, 4);
    cvtT(Wo, wbf + oWo, DMODEL, DMODEL, 4);
    cvtT(fc1cW, wbf + oF1c, DMODEL, DFFN, 4);
    cvtT(fc2cW, wbf + oF2c, DFFN, DMODEL, 4);
    cvtT(fc1vW, wbf + oF1v, DMODEL, DFFN, 4);
    cvtT(fc2vW, wbf + oF2v, DFFN, DMODEL, 4);
    cvtT(hW1, wbf + oH1, 1536, DFFN, 1);
    cvtT(hW2, wbf + oH2, DFFN, 256, 1);

    embed_kernel<<<MROWS, 256, 0, stream>>>(q_ids, r_ids, ctx_emb, val_emb, pos_emb,
                                            ctxf, ctxb, valf, valb);

    auto gemm = [&](const unsigned short* A, const unsigned short* Bt, const float* bias,
                    float* Cf, unsigned short* Cb, int M, int N, int K, int act) {
        wmma_gemm_kernel<<<dim3(N / 128, M / 128), 256, 0, stream>>>(A, Bt, bias, Cf, Cb, M, N, K, act);
    };

    for (int i = 0; i < 4; ++i) {
        const size_t wD = (size_t)i * DMODEL * DMODEL;   // 262144
        const size_t wF = (size_t)i * DMODEL * DFFN;     // 1048576
        gemm(ctxb, wbf + oWq + wD, bq + i * DMODEL, nullptr, qb, MROWS, DMODEL, DMODEL, 0);
        gemm(ctxb, wbf + oWk + wD, bk + i * DMODEL, nullptr, kb, MROWS, DMODEL, DMODEL, 0);
        gemm(valb, wbf + oWv + wD, bv + i * DMODEL, nullptr, vb, MROWS, DMODEL, DMODEL, 0);
        vtrans_kernel<<<MROWS, 256, 0, stream>>>(vb, vt);
        attn_kernel<<<dim3(LSEQ / 64, NHEAD, BDIM), 128, 0, stream>>>(qb, kb, vt, attb);
        gemm(attb, wbf + oWo + wD, bo + i * DMODEL, tmpf, nullptr, MROWS, DMODEL, DMODEL, 0);
        ln_residual_kernel<<<MROWS, 256, 0, stream>>>(ctxf, tmpf, ln1c_s + i * DMODEL,
                                                      ln1c_b + i * DMODEL, ctxf, ctxb);
        ln_residual_kernel<<<MROWS, 256, 0, stream>>>(valf, tmpf, ln1v_s + i * DMODEL,
                                                      ln1v_b + i * DMODEL, valf, valb);
        gemm(ctxb, wbf + oF1c + wF, fc1cb + i * DFFN,   nullptr, bigb, MROWS, DFFN, DMODEL, 1);
        gemm(bigb, wbf + oF2c + wF, fc2cb + i * DMODEL, tmpf, nullptr, MROWS, DMODEL, DFFN, 0);
        ln_residual_kernel<<<MROWS, 256, 0, stream>>>(ctxf, tmpf, ln2c_s + i * DMODEL,
                                                      ln2c_b + i * DMODEL, ctxf, ctxb);
        gemm(valb, wbf + oF1v + wF, fc1vb + i * DFFN,   nullptr, bigb, MROWS, DFFN, DMODEL, 1);
        gemm(bigb, wbf + oF2v + wF, fc2vb + i * DMODEL, tmpf, nullptr, MROWS, DMODEL, DFFN, 0);
        ln_residual_kernel<<<MROWS, 256, 0, stream>>>(valf, tmpf, ln2v_s + i * DMODEL,
                                                      ln2v_b + i * DMODEL, valf, valb);
    }

    feat_kernel<<<MROWS, 256, 0, stream>>>(ctxf, valf, skill, q_ids, featb);
    gemm(featb, wbf + oH1, hb1, nullptr, bigb, MROWS, DFFN, 1536, 1);
    gemm(bigb,  wbf + oH2, hb2, h2f, nullptr, MROWS, 256, DFFN, 1);
    head_final_kernel<<<MROWS / 8, 256, 0, stream>>>(h2f, hW3, hb3, (float*)d_out);
    (void)in_sizes; (void)n_in; (void)out_size; (void)ws_size;
}
